// TransGIN_24859270709380
// MI455X (gfx1250) — matrix operand order repaired and checked by reading, not compiled, not run on hardware
//
#include <hip/hip_runtime.h>
#include <stdint.h>

// ---------------------------------------------------------------------------
// TransGIN forward for MI455X (gfx1250), wave32 + WMMA bf16 (16x16x32).
// All WMMA operands are bf16 at rest: weights pre-transposed bf16,
// activations feeding GEMM A-sides shadowed in bf16, V pre-transposed.
// Inner GEMM loops = b128 loads + v_wmma only. fp32 kept for softmax /
// layernorm / residual numerics.
// ---------------------------------------------------------------------------

typedef __attribute__((ext_vector_type(16))) __bf16          v16bf;
typedef __attribute__((ext_vector_type(16))) unsigned short  v16u;
typedef __attribute__((ext_vector_type(8)))  unsigned short  v8u;
typedef __attribute__((ext_vector_type(8)))  float           v8f;

#define BNS 0.9999950000374996f   // 1/sqrt(1+1e-5)

__device__ __forceinline__ unsigned short f2bf(float f) {
  union { float f; unsigned u; } x; x.f = f;
  unsigned u = x.u;
  unsigned r = u + 0x7FFFu + ((u >> 16) & 1u);   // round-to-nearest-even
  return (unsigned short)(r >> 16);
}
__device__ __forceinline__ float bf2f(unsigned short s) {
  union { unsigned u; float f; } x; x.u = ((unsigned)s) << 16;
  return x.f;
}
__device__ __forceinline__ v16bf bits2bf(v16u u) {
  return __builtin_bit_cast(v16bf, u);
}
__device__ __forceinline__ v8f wmma_bf16(v16bf a, v16bf b, v8f c) {
  return __builtin_amdgcn_wmma_f32_16x16x32_bf16(false, a, false, b, (short)0, c,
                                                 false, false);
}

// ---- fragment loaders (all bf16) ------------------------------------------
// A fragment (16x32): p = &A[row*lda + k0 + hh*16]; elems 0..15 contiguous at p.
__device__ __forceinline__ v16bf load_a_bf16(const unsigned short* __restrict__ p) {
  v8u lo = *reinterpret_cast<const v8u*>(p);
  v8u hi = *reinterpret_cast<const v8u*>(p + 8);
  v16u u = __builtin_shufflevector(lo, hi, 0,1,2,3,4,5,6,7,8,9,10,11,12,13,14,15);
  return bits2bf(u);
}
// B fragment (32x16): 16 contiguous bf16 (two b128 loads).
__device__ __forceinline__ v16bf load_b_bf16_contig(const unsigned short* __restrict__ p) {
  v16u u = *reinterpret_cast<const v16u*>(p);
  return bits2bf(u);
}
// B fragment, column-strided bf16 (small GIN aggregation GEMM only).
__device__ __forceinline__ v16bf load_b_bf16_strided(const unsigned short* __restrict__ p,
                                                     size_t stride) {
  v16u u;
#pragma unroll
  for (int e = 0; e < 16; ++e) u[e] = p[(size_t)e * stride];
  return bits2bf(u);
}

// ---------------------------------------------------------------------------
// Weight prep: W[K][N] fp32 -> WT[N][K] bf16
// ---------------------------------------------------------------------------
__global__ void wconv_t_kernel(const float* __restrict__ W, unsigned short* __restrict__ WT,
                               int K, int N)
{
  size_t idx = (size_t)blockIdx.x * blockDim.x + threadIdx.x;
  if (idx >= (size_t)K * N) return;
  int n = (int)(idx / K);
  int k = (int)(idx - (size_t)n * K);
  WT[idx] = f2bf(W[(size_t)k * N + n]);
}

// ---------------------------------------------------------------------------
// GEMM: C[M,N] = epilogue(A[M,K] @ B[K,N]); A bf16, B pre-transposed bf16.
// CBF: C written bf16. M%64==0, N%64==0, K%32==0.
// epilogue: +bias; bnorder==1 bn-before-act; act=lrelu; bnorder==2 bn-after.
// ---------------------------------------------------------------------------
template<int CBF>
__global__ void __launch_bounds__(128) gemm_tb(
    const unsigned short* __restrict__ A, const unsigned short* __restrict__ BT,
    void* __restrict__ Cp, int M, int N, int K,
    const float* __restrict__ bias, const float* __restrict__ scale,
    const float* __restrict__ shift, int bnorder, int act)
{
  const int lane = threadIdx.x & 31;
  const int wave = threadIdx.x >> 5;
  const int half = lane >> 4;
  const int l16  = lane & 15;
  const int n0 = blockIdx.x * 64;
  const int m0 = blockIdx.y * 64 + wave * 16;

  v8f z = {0.f,0.f,0.f,0.f,0.f,0.f,0.f,0.f};
  v8f acc[4] = {z, z, z, z};

  const int arow = m0 + l16;
  for (int k0 = 0; k0 < K; k0 += 32) {
    v16bf a = load_a_bf16(A + (size_t)arow * K + k0 + half * 16);
    const int kb = k0 + half * 16;
#pragma unroll
    for (int t = 0; t < 4; ++t) {
      const int n = n0 + t * 16 + l16;
      v16bf b = load_b_bf16_contig(BT + (size_t)n * K + kb);
      acc[t] = wmma_bf16(a, b, acc[t]);
    }
  }
#pragma unroll
  for (int t = 0; t < 4; ++t) {
    const int n = n0 + t * 16 + l16;
    const float bi = bias  ? bias[n]        : 0.f;
    const float sc = scale ? scale[n] * BNS : 1.f;
    const float sh = shift ? shift[n]       : 0.f;
#pragma unroll
    for (int i = 0; i < 8; ++i) {
      const int row = m0 + half * 8 + i;
      float v = acc[t][i] + bi;
      if (bnorder == 1) v = v * sc + sh;
      if (act)          v = (v >= 0.f) ? v : 0.01f * v;
      if (bnorder == 2) v = v * sc + sh;
      if (CBF) ((unsigned short*)Cp)[(size_t)row * N + n] = f2bf(v);
      else     ((float*)Cp)[(size_t)row * N + n] = v;
    }
  }
}

// VT[b][h][dh][s] = qkv[b][s][512 + h*32 + dh]  (bf16 -> bf16 transpose)
__global__ void __launch_bounds__(256) vt_kernel(
    const unsigned short* __restrict__ qkv, unsigned short* __restrict__ vt)
{
  size_t idx = (size_t)blockIdx.x * 256 + threadIdx.x;   // 1024*8*32*64
  int s  = (int)(idx & 63);
  int dh = (int)((idx >> 6) & 31);
  int h  = (int)((idx >> 11) & 7);
  int b  = (int)(idx >> 14);
  vt[idx] = qkv[((size_t)b * 64 + s) * 768 + 512 + h * 32 + dh];
}

// ---------------------------------------------------------------------------
// Attention: 1 block / batch, 1 wave / head. S=64, DH=32, H=8.
// qkv bf16 [b][s][768]; vt bf16 [b][h][32][64]. Writes ctx bf16 [b][s][256]
// and attnmean[b][64][64] fp32 (mean over heads).
// ---------------------------------------------------------------------------
__global__ void __launch_bounds__(256) attention_kernel(
    const unsigned short* __restrict__ qkv, const unsigned short* __restrict__ vt,
    unsigned short* __restrict__ ctx, float* __restrict__ amean)
{
  __shared__ unsigned short sAttn[8 * 64 * 64];   // 64 KB bf16 attn
  const int b    = blockIdx.x;
  const int tid  = threadIdx.x;
  const int h    = tid >> 5;
  const int lane = tid & 31;
  const int half = lane >> 4;
  const int l16  = lane & 15;
  const size_t qb = (size_t)b * 64 * 768;

  v8f z = {0.f,0.f,0.f,0.f,0.f,0.f,0.f,0.f};

  // ---- scores = Q @ K^T (64x64, K=32) ----
  v8f acc[4][4];
#pragma unroll
  for (int a1 = 0; a1 < 4; ++a1)
#pragma unroll
    for (int a2 = 0; a2 < 4; ++a2) acc[a1][a2] = z;

  v16bf afr[4];
#pragma unroll
  for (int ti = 0; ti < 4; ++ti) {
    const int s = ti * 16 + l16;
    afr[ti] = load_a_bf16(qkv + qb + (size_t)s * 768 + h * 32 + half * 16);
  }
#pragma unroll
  for (int tj = 0; tj < 4; ++tj) {
    const int n = tj * 16 + l16;          // key index
    v16bf bfr = load_b_bf16_contig(qkv + qb + (size_t)n * 768 + 256 + h * 32 + half * 16);
#pragma unroll
    for (int ti = 0; ti < 4; ++ti) acc[ti][tj] = wmma_bf16(afr[ti], bfr, acc[ti][tj]);
  }

  // ---- softmax rows (scale 1/sqrt(32)); a row lives in one 16-lane half ----
  const float scal = 0.17677669529663687f;
#pragma unroll
  for (int ti = 0; ti < 4; ++ti) {
#pragma unroll
    for (int i = 0; i < 8; ++i) {
      float mx = -3.4e38f;
#pragma unroll
      for (int tj = 0; tj < 4; ++tj) {
        acc[ti][tj][i] *= scal;
        mx = fmaxf(mx, acc[ti][tj][i]);
      }
      for (int o = 1; o < 16; o <<= 1) mx = fmaxf(mx, __shfl_xor(mx, o, 32));
      float sum = 0.f;
#pragma unroll
      for (int tj = 0; tj < 4; ++tj) {
        float e = __expf(acc[ti][tj][i] - mx);
        acc[ti][tj][i] = e;
        sum += e;
      }
      for (int o = 1; o < 16; o <<= 1) sum += __shfl_xor(sum, o, 32);
      float r = 1.0f / sum;
#pragma unroll
      for (int tj = 0; tj < 4; ++tj) acc[ti][tj][i] *= r;
    }
  }

  // ---- stash attn (bf16) in LDS ----
#pragma unroll
  for (int ti = 0; ti < 4; ++ti)
#pragma unroll
    for (int tj = 0; tj < 4; ++tj)
#pragma unroll
      for (int i = 0; i < 8; ++i) {
        const int row = ti * 16 + half * 8 + i;
        const int col = tj * 16 + l16;
        sAttn[(h * 64 + row) * 64 + col] = f2bf(acc[ti][tj][i]);
      }
  __syncthreads();

  // ---- attn mean over heads -> global ----
  for (int idx = tid; idx < 4096; idx += 256) {
    float s = 0.f;
#pragma unroll
    for (int hh = 0; hh < 8; ++hh) s += bf2f(sAttn[hh * 4096 + idx]);
    amean[(size_t)b * 4096 + idx] = s * 0.125f;
  }

  // ---- ctx = attn @ V (64x32, K=64), V pre-transposed: contiguous B frags ----
  v8f cacc[4][2];
#pragma unroll
  for (int a1 = 0; a1 < 4; ++a1) { cacc[a1][0] = z; cacc[a1][1] = z; }

  for (int k0 = 0; k0 < 64; k0 += 32) {
    v16bf af[4];
#pragma unroll
    for (int ti = 0; ti < 4; ++ti) {
      const unsigned short* ap =
          &sAttn[(h * 64 + ti * 16 + l16) * 64 + k0 + half * 16];
      v16u u;
#pragma unroll
      for (int e = 0; e < 8; ++e) u[e] = ap[e];
#pragma unroll
      for (int e = 0; e < 8; ++e) u[8 + e] = ap[8 + e];
      af[ti] = bits2bf(u);
    }
#pragma unroll
    for (int t = 0; t < 2; ++t) {
      const int dh = t * 16 + l16;
      const unsigned short* vp =
          vt + (((size_t)b * 8 + h) * 32 + dh) * 64 + k0 + half * 16;
      v16bf bf = load_b_bf16_contig(vp);
#pragma unroll
      for (int ti = 0; ti < 4; ++ti) cacc[ti][t] = wmma_bf16(af[ti], bf, cacc[ti][t]);
    }
  }
#pragma unroll
  for (int ti = 0; ti < 4; ++ti)
#pragma unroll
    for (int t = 0; t < 2; ++t)
#pragma unroll
      for (int i = 0; i < 8; ++i) {
        const int s  = ti * 16 + half * 8 + i;
        const int dh = t * 16 + l16;
        ctx[((size_t)b * 64 + s) * 256 + h * 32 + dh] = f2bf(cacc[ti][t][i]);
      }
}

// ---------------------------------------------------------------------------
// x = LayerNorm(x + r) * g + b (fp32), plus bf16 shadow copy for GEMM A-sides.
// ---------------------------------------------------------------------------
__global__ void __launch_bounds__(256) residual_ln_kernel(
    float* __restrict__ x, unsigned short* __restrict__ xb, const float* __restrict__ r,
    const float* __restrict__ g, const float* __restrict__ bb)
{
  __shared__ float red[256];
  const size_t row = blockIdx.x;
  const int d = threadIdx.x;
  float v = x[row * 256 + d] + r[row * 256 + d];
  red[d] = v;
  __syncthreads();
  for (int o = 128; o > 0; o >>= 1) { if (d < o) red[d] += red[d + o]; __syncthreads(); }
  const float mean = red[0] * (1.f / 256.f);
  __syncthreads();
  const float c = v - mean;
  red[d] = c * c;
  __syncthreads();
  for (int o = 128; o > 0; o >>= 1) { if (d < o) red[d] += red[d + o]; __syncthreads(); }
  const float var = red[0] * (1.f / 256.f);
  const float y = c * rsqrtf(var + 1e-5f) * g[d] + bb[d];
  x[row * 256 + d]  = y;
  xb[row * 256 + d] = f2bf(y);
}

__global__ void mask_update_kernel(float* __restrict__ mask, const float* __restrict__ amean,
                                   const float* __restrict__ thr, int n)
{
  int i = blockIdx.x * blockDim.x + threadIdx.x;
  if (i < n) {
    float v = (amean[i] > thr[0]) ? 1.f : 0.f;
    mask[i] = fmaxf(mask[i], v);
  }
}

// Aeff[b] = I + (mask | mask^T), bf16, feeds WMMA A fragments.
__global__ void adj_build_kernel(const float* __restrict__ mask,
                                 unsigned short* __restrict__ aeff)
{
  int idx = blockIdx.x * blockDim.x + threadIdx.x;
  if (idx >= 1024 * 4096) return;
  int b = idx >> 12;
  int ij = idx & 4095;
  int i = ij >> 6, j = ij & 63;
  size_t base = (size_t)b << 12;
  float m = ((mask[base + ij] > 0.f) || (mask[base + (j << 6) + i] > 0.f)) ? 1.f : 0.f;
  aeff[idx] = f2bf(m + ((i == j) ? 1.f : 0.f));
}

// agg[b] (bf16) = Aeff[b] (64x64 bf16) @ h[b] (64xF bf16). Covers h + adj@h.
__global__ void __launch_bounds__(128) gin_agg_kernel(
    const unsigned short* __restrict__ aeff, const unsigned short* __restrict__ h,
    unsigned short* __restrict__ agg, int F)
{
  const int lane = threadIdx.x & 31;
  const int wave = threadIdx.x >> 5;
  const int half = lane >> 4;
  const int l16  = lane & 15;
  const int b  = blockIdx.y;
  const int n0 = blockIdx.x * 64;

  v8f z = {0.f,0.f,0.f,0.f,0.f,0.f,0.f,0.f};
  v8f acc[4] = {z, z, z, z};

  const int row = wave * 16 + l16;
  for (int k0 = 0; k0 < 64; k0 += 32) {
    v16bf a = load_a_bf16(aeff + ((size_t)b << 12) + row * 64 + k0 + half * 16);
    const int kb = k0 + half * 16;
#pragma unroll
    for (int t = 0; t < 4; ++t) {
      const unsigned short* bp = h + ((size_t)b * 64 + kb) * F + n0 + t * 16 + l16;
      acc[t] = wmma_bf16(a, load_b_bf16_strided(bp, (size_t)F), acc[t]);
    }
  }
#pragma unroll
  for (int t = 0; t < 4; ++t)
#pragma unroll
    for (int i = 0; i < 8; ++i)
      agg[((size_t)b * 64 + wave * 16 + half * 8 + i) * F + n0 + t * 16 + l16] =
          f2bf(acc[t][i]);
}

// h (bf16) = lrelu(bn_eval(m2 bf16, g, b)), feature dim 1024.
__global__ void bn_lrelu_kernel(const unsigned short* __restrict__ in,
                                unsigned short* __restrict__ out,
                                const float* __restrict__ g, const float* __restrict__ bb,
                                size_t n)
{
  size_t i = (size_t)blockIdx.x * blockDim.x + threadIdx.x;
  if (i < n) {
    int f = (int)(i & 1023);
    float v = bf2f(in[i]) * (g[f] * BNS) + bb[f];
    out[i] = f2bf((v >= 0.f) ? v : 0.01f * v);
  }
}

// hcat[b][slot*1024 + f] = sum_s h[b][s][f]  (fp32 accumulate, bf16 store)
__global__ void pool_kernel(const unsigned short* __restrict__ h,
                            unsigned short* __restrict__ hcat, int slot)
{
  int b = blockIdx.x;
  int f = blockIdx.y * 256 + threadIdx.x;
  float s = 0.f;
#pragma unroll 4
  for (int i = 0; i < 64; ++i) s += bf2f(h[((size_t)b * 64 + i) * 1024 + f]);
  hcat[(size_t)b * 3136 + slot * 1024 + f] = f2bf(s);
}

__global__ void kmers_kernel(const float* __restrict__ src0, unsigned short* __restrict__ hcat)
{
  int b = blockIdx.x;
  int j = threadIdx.x;   // 64 threads
  hcat[(size_t)b * 3136 + 3072 + j] = f2bf(src0[(size_t)b * 128 + 64 + j]);
}

// x[b][s][d] = emb[tok(b,s)][d] + PE[s][d]  (fp32 + bf16 shadow)
__global__ void __launch_bounds__(256) embed_pe_kernel(
    const float* __restrict__ src0, const float* __restrict__ emb,
    float* __restrict__ x, unsigned short* __restrict__ xb)
{
  int bs = blockIdx.x;
  int d  = threadIdx.x;
  int b = bs >> 6, s = bs & 63;
  int tok = (int)src0[(size_t)b * 128 + s];
  int i = d >> 1;
  float freq = __expf(-9.210340371976184f * (float)(2 * i) * (1.f / 256.f)); // ln(1e4)
  float ang  = (float)s * freq;
  float pe   = (d & 1) ? __cosf(ang) : __sinf(ang);
  float v = emb[(size_t)tok * 256 + d] + pe;
  x[(size_t)bs * 256 + d]  = v;
  xb[(size_t)bs * 256 + d] = f2bf(v);
}

// out[b][c] = h1[b][:] @ lin2_w[:,c] + lin2_b[c]   (K=64, N=2)
__global__ void lin2_kernel(const float* __restrict__ h1, const float* __restrict__ w,
                            const float* __restrict__ bias, float* __restrict__ out)
{
  int idx = blockIdx.x * blockDim.x + threadIdx.x;
  if (idx >= 2048) return;
  int b = idx >> 1, c = idx & 1;
  float s = bias[c];
#pragma unroll 8
  for (int k = 0; k < 64; ++k) s += h1[b * 64 + k] * w[k * 2 + c];
  out[idx] = s;
}

// ---------------------------------------------------------------------------
// Orchestration
// ---------------------------------------------------------------------------
extern "C" void kernel_launch(void* const* d_in, const int* in_sizes, int n_in,
                              void* d_out, int out_size, void* d_ws, size_t ws_size,
                              hipStream_t stream)
{
  (void)in_sizes; (void)n_in; (void)out_size; (void)ws_size;
  const float* src0 = (const float*)d_in[0];
  const float* thr  = (const float*)d_in[1];
  const float* emb  = (const float*)d_in[2];
  auto TL = [&](int L, int j) -> const float* { return (const float*)d_in[3 + L * 12 + j]; };
  auto GL = [&](int g, int j) -> const float* { return (const float*)d_in[27 + g * 8 + j]; };
  const float* lin1_w = (const float*)d_in[51];
  const float* lin1_b = (const float*)d_in[52];
  const float* bn1_g  = (const float*)d_in[53];
  const float* bn1_b  = (const float*)d_in[54];
  const float* lin2_w = (const float*)d_in[55];
  const float* lin2_b = (const float*)d_in[56];
  float* out = (float*)d_out;

  // ---- workspace arena (byte offsets, 256B aligned) ----
  char* base = (char*)d_ws;
  size_t off = 0;
  auto alloc = [&](size_t bytes) -> void* {
    off = (off + 255) & ~(size_t)255;
    void* p = base + off;
    off += bytes;
    return p;
  };
  float*          X     = (float*)alloc((size_t)65536 * 256 * 4);            // residual stream
  unsigned short* XB    = (unsigned short*)alloc((size_t)65536 * 256 * 2);   // bf16 shadow of X
  unsigned short* ACT1  = (unsigned short*)alloc((size_t)65536 * 1024 * 2);  // qkv | agg | m2
  unsigned short* ACT2  = (unsigned short*)alloc((size_t)65536 * 1024 * 2);  // ff1 | m
  unsigned short* CTXB  = (unsigned short*)alloc((size_t)65536 * 256 * 2);   // attention ctx
  unsigned short* VT    = (unsigned short*)alloc((size_t)65536 * 256 * 2);   // V transposed
  float*          SRC2  = (float*)alloc((size_t)65536 * 256 * 4);
  float*          FF2   = (float*)alloc((size_t)65536 * 256 * 4);
  unsigned short* HB    = (unsigned short*)alloc((size_t)65536 * 1024 * 2);  // GIN h (bf16)
  float*          MASK  = (float*)alloc((size_t)4194304 * 4);
  float*          AMEAN = (float*)alloc((size_t)4194304 * 4);
  unsigned short* AEFF  = (unsigned short*)alloc((size_t)4194304 * 2);
  unsigned short* HCATB = (unsigned short*)alloc((size_t)1024 * 3136 * 2);
  float*          H1    = (float*)alloc((size_t)1024 * 64 * 4);
  // bf16 transposed weights
  unsigned short* wqkvT[2]; unsigned short* woT[2];
  unsigned short* w1T[2];   unsigned short* w2T[2];
  unsigned short* waT[3];   unsigned short* wbT[3];
  for (int L = 0; L < 2; ++L) {
    wqkvT[L] = (unsigned short*)alloc((size_t)256 * 768 * 2);
    woT[L]   = (unsigned short*)alloc((size_t)256 * 256 * 2);
    w1T[L]   = (unsigned short*)alloc((size_t)256 * 1024 * 2);
    w2T[L]   = (unsigned short*)alloc((size_t)1024 * 256 * 2);
  }
  for (int g = 0; g < 3; ++g) {
    waT[g] = (unsigned short*)alloc((size_t)(g == 0 ? 256 : 1024) * 1024 * 2);
    wbT[g] = (unsigned short*)alloc((size_t)1024 * 1024 * 2);
  }
  unsigned short* lin1T = (unsigned short*)alloc((size_t)3136 * 64 * 2);

  auto wconv = [&](const float* W, unsigned short* WT, int K, int N) {
    size_t n = (size_t)K * N;
    wconv_t_kernel<<<(unsigned)((n + 255) / 256), 256, 0, stream>>>(W, WT, K, N);
  };
  auto gemm = [&](const unsigned short* A, const unsigned short* BT, void* C,
                  int M, int N, int K, const float* bias, const float* scale,
                  const float* shift, int bnorder, int act, int cbf) {
    dim3 grid((unsigned)(N / 64), (unsigned)(M / 64));
    if (cbf)
      gemm_tb<1><<<grid, 128, 0, stream>>>(A, BT, C, M, N, K, bias, scale, shift, bnorder, act);
    else
      gemm_tb<0><<<grid, 128, 0, stream>>>(A, BT, C, M, N, K, bias, scale, shift, bnorder, act);
  };

  // ---- weight prep (cheap vs ~1 TFLOP of GEMM work) ----
  for (int L = 0; L < 2; ++L) {
    wconv(TL(L,0), wqkvT[L], 256, 768);
    wconv(TL(L,2), woT[L],   256, 256);
    wconv(TL(L,4), w1T[L],   256, 1024);
    wconv(TL(L,6), w2T[L],   1024, 256);
  }
  for (int g = 0; g < 3; ++g) {
    wconv(GL(g,0), waT[g], (g == 0 ? 256 : 1024), 1024);
    wconv(GL(g,4), wbT[g], 1024, 1024);
  }
  wconv(lin1_w, lin1T, 3136, 64);

  hipMemsetAsync(MASK, 0, (size_t)4194304 * sizeof(float), stream);

  embed_pe_kernel<<<65536, 256, 0, stream>>>(src0, emb, X, XB);

  const int M = 65536;   // B*S
  for (int L = 0; L < 2; ++L) {
    gemm(XB, wqkvT[L], ACT1, M, 768, 256, TL(L,1), nullptr, nullptr, 0, 0, 1); // qkv bf16
    vt_kernel<<<65536, 256, 0, stream>>>(ACT1, VT);
    attention_kernel<<<1024, 256, 0, stream>>>(ACT1, VT, CTXB, AMEAN);
    mask_update_kernel<<<(4194304 + 255) / 256, 256, 0, stream>>>(MASK, AMEAN, thr, 4194304);
    gemm(CTXB, woT[L], SRC2, M, 256, 256, TL(L,3), nullptr, nullptr, 0, 0, 0); // Wo
    residual_ln_kernel<<<65536, 256, 0, stream>>>(X, XB, SRC2, TL(L,8), TL(L,9));
    gemm(XB, w1T[L], ACT2, M, 1024, 256, TL(L,5), nullptr, nullptr, 0, 1, 1);  // W1+lrelu
    gemm(ACT2, w2T[L], FF2, M, 256, 1024, TL(L,7), nullptr, nullptr, 0, 0, 0); // W2
    residual_ln_kernel<<<65536, 256, 0, stream>>>(X, XB, FF2, TL(L,10), TL(L,11));
  }

  adj_build_kernel<<<(4194304 + 255) / 256, 256, 0, stream>>>(MASK, AEFF);

  const unsigned short* hin = XB;
  for (int g = 0; g < 3; ++g) {
    const int F = (g == 0) ? 256 : 1024;
    gin_agg_kernel<<<dim3((unsigned)(F / 64), 1024), 128, 0, stream>>>(AEFF, hin, ACT1, F);
    // m = lrelu(bn(agg@Wa + ba))  -> bf16
    gemm(ACT1, waT[g], ACT2, M, 1024, F, GL(g,1), GL(g,2), GL(g,3), 1, 1, 1);
    // m2 = lrelu(m@Wb + bb)       -> bf16
    gemm(ACT2, wbT[g], ACT1, M, 1024, 1024, GL(g,5), nullptr, nullptr, 0, 1, 1);
    // h = lrelu(bn(m2))           -> bf16
    bn_lrelu_kernel<<<(67108864 + 255) / 256, 256, 0, stream>>>(ACT1, HB, GL(g,6), GL(g,7),
                                                                (size_t)67108864);
    pool_kernel<<<dim3(1024, 4), 256, 0, stream>>>(HB, HCATB, g);
    hin = HB;
  }

  kmers_kernel<<<1024, 64, 0, stream>>>(src0, HCATB);
  // h1 = bn(lrelu(hcat@lin1 + b))
  gemm(HCATB, lin1T, H1, 1024, 64, 3136, lin1_b, bn1_g, bn1_b, 2, 1, 0);
  lin2_kernel<<<8, 256, 0, stream>>>(H1, lin2_w, lin2_b, out);
}
